// CrossAttention_31688268710398
// MI455X (gfx1250) — matrix operand order repaired and checked by reading, not compiled, not run on hardware
//
#include <hip/hip_runtime.h>
#include <stdint.h>

#define B_  2
#define LQ_ 2048
#define LK_ 2048
#define E_  1024
#define H_  16
#define D_  64

typedef __attribute__((ext_vector_type(16))) __bf16 v16bf;
typedef __attribute__((ext_vector_type(8)))  float  v8f;

union FragBF {
    v16bf v;
    uint4 q[2];
    unsigned short h[16];
};

__device__ __forceinline__ unsigned short f2bf(float f) {
    union { float f; unsigned u; } x;
    x.f = f;
    unsigned r = x.u + 0x7FFFu + ((x.u >> 16) & 1u);  // round-to-nearest-even
    return (unsigned short)(r >> 16);
}

// LDS byte offset of a __shared__ object (addrspace(3) pointers are 32-bit offsets)
__device__ __forceinline__ unsigned lds_off(const void* p) {
    return (unsigned)(unsigned long long)
        (__attribute__((address_space(3))) const void*)p;
}

// GVS-form async copy: LDS[vdst] = MEM[saddr + voffset]  (16 bytes, ASYNCcnt)
__device__ __forceinline__ void async_ld_b128(unsigned ldsaddr, unsigned goff,
                                              const void* base) {
    asm volatile("global_load_async_to_lds_b128 %0, %1, %2"
                 :: "v"(ldsaddr), "v"(goff), "s"(base) : "memory");
}
__device__ __forceinline__ void wait_async0() {
    asm volatile("s_wait_asynccnt 0x0" ::: "memory");
}

// ---------------------------------------------------------------- convert
__global__ void cvt_f32_bf16(const float* __restrict__ src,
                             unsigned short* __restrict__ dst, int n) {
    for (int i = blockIdx.x * blockDim.x + threadIdx.x; i < n;
         i += gridDim.x * blockDim.x)
        dst[i] = f2bf(src[i]);
}

// ---------------------------------------------------------------- GEMM: X @ W^T + b
// Workgroup (8 waves) computes a 128(M) x 64(O) tile; K double-buffered through
// LDS via global_load_async_to_lds_b128. Each wave: 16 rows x 64 cols (4 accs).
// out_mode 0: bf16 head layout [b][h][l][d]
// out_mode 1: bf16 V-transposed [b][h][d][l]
// out_mode 2: f32 row-major (N, E_)
__global__ void gemm_bf16(const unsigned short* __restrict__ A,
                          const unsigned short* __restrict__ W,
                          const float* __restrict__ bias,
                          unsigned short* __restrict__ out_bf,
                          float* __restrict__ out_f32,
                          int out_mode)
{
    __shared__ unsigned short As[2][128 * 32];  // 16 KB
    __shared__ unsigned short Bs[2][64 * 32];   //  8 KB

    const int tid  = threadIdx.x;
    const int lane = tid & 31;
    const int wave = tid >> 5;
    const int OT = E_ / 64;                     // 16 col-tiles
    const int m0 = (blockIdx.x / OT) * 128;
    const int o0 = (blockIdx.x % OT) * 64;
    const int kh = lane >> 4;
    const int ml = lane & 15;

    const unsigned asb[2] = { lds_off(&As[0][0]), lds_off(&As[1][0]) };
    const unsigned bsb[2] = { lds_off(&Bs[0][0]), lds_off(&Bs[1][0]) };

    // ---- async stage of one K-slab (32 wide) into LDS buffer `buf`
    auto fill = [&](int buf, int kk) {
        // A tile: 128 rows x 32 cols bf16 = 512 x 16B chunks, 2 per thread
        #pragma unroll
        for (int i = 0; i < 2; ++i) {
            const int c = tid + 256 * i;
            const int row = c >> 2, col = (c & 3) * 8;
            async_ld_b128(asb[buf] + (unsigned)(row * 32 + col) * 2u,
                          (unsigned)(((m0 + row) * E_ + kk + col) * 2), A);
        }
        // B tile: 64 rows x 32 cols = 256 x 16B chunks, 1 per thread
        {
            const int row = tid >> 2, col = (tid & 3) * 8;
            async_ld_b128(bsb[buf] + (unsigned)(row * 32 + col) * 2u,
                          (unsigned)(((o0 + row) * E_ + kk + col) * 2), W);
        }
    };

    fill(0, 0);
    wait_async0();
    __syncthreads();

    v8f acc[4] = {};
    for (int kk = 0; kk < E_; kk += 32) {
        const int cur = (kk >> 5) & 1;
        if (kk + 32 < E_) fill(cur ^ 1, kk + 32);

        FragBF a;
        const unsigned short* ar = &As[cur][(16 * wave + ml) * 32];
        a.q[0] = *(const uint4*)(ar + 8 * kh);
        a.q[1] = *(const uint4*)(ar + 16 + 8 * kh);
        #pragma unroll
        for (int j = 0; j < 4; ++j) {
            FragBF b;
            const unsigned short* br = &Bs[cur][(16 * j + ml) * 32];
            b.q[0] = *(const uint4*)(br + 8 * kh);
            b.q[1] = *(const uint4*)(br + 16 + 8 * kh);
            acc[j] = __builtin_amdgcn_wmma_f32_16x16x32_bf16(
                false, a.v, false, b.v, (short)0, acc[j], false, false);
        }
        wait_async0();
        __syncthreads();
    }

    const int n = lane & 15;
    #pragma unroll
    for (int j = 0; j < 4; ++j) {
        const int o = o0 + 16 * j + n;
        const float bofs = bias[o];
        #pragma unroll
        for (int r = 0; r < 8; ++r) {
            const int m = m0 + 16 * wave + r + 8 * (lane >> 4);
            float val = acc[j][r] + bofs;
            if (out_mode == 2) {
                out_f32[(size_t)m * E_ + o] = val;
            } else {
                const int b_ = m / LQ_, l = m % LQ_;
                const int hh = o / D_,  d = o % D_;
                size_t idx;
                if (out_mode == 0) idx = ((size_t)(b_ * H_ + hh) * LQ_ + l) * D_ + d;
                else               idx = ((size_t)(b_ * H_ + hh) * D_  + d) * LK_ + l;
                out_bf[idx] = f2bf(val);
            }
        }
    }
}

// ---------------------------------------------------------------- scores = scale * Q @ K^T
// One wave -> 16(q) x 64(k) block: Q fragment reused across 4 accumulators.
__global__ void scores_kernel(const unsigned short* __restrict__ Qb,  // [bh][lq][d]
                              const unsigned short* __restrict__ Kb,  // [bh][lk][d]
                              float* __restrict__ attn)               // [bh][lq][lk]
{
    const int lane = threadIdx.x & 31;
    const int wave = threadIdx.x >> 5;
    const int tile = blockIdx.x * (blockDim.x >> 5) + wave;
    const int tilesPerBH = (LQ_ / 16) * (LK_ / 64);     // 128 * 32
    const int bh = tile / tilesPerBH;
    const int rr = tile % tilesPerBH;
    const int q0 = (rr / (LK_ / 64)) * 16;
    const int k0 = (rr % (LK_ / 64)) * 64;
    const int kh = lane >> 4;
    const int ml = lane & 15;
    const unsigned short* qrow = Qb + ((size_t)bh * LQ_ + q0 + ml) * D_;

    v8f acc[4] = {};
    #pragma unroll
    for (int d0 = 0; d0 < D_; d0 += 32) {
        FragBF a;
        a.q[0] = *(const uint4*)(qrow + d0 + 8 * kh);
        a.q[1] = *(const uint4*)(qrow + d0 + 16 + 8 * kh);
        #pragma unroll
        for (int j = 0; j < 4; ++j) {
            const unsigned short* krow =
                Kb + ((size_t)bh * LK_ + k0 + 16 * j + ml) * D_;
            FragBF b;
            b.q[0] = *(const uint4*)(krow + d0 + 8 * kh);
            b.q[1] = *(const uint4*)(krow + d0 + 16 + 8 * kh);
            acc[j] = __builtin_amdgcn_wmma_f32_16x16x32_bf16(
                false, a.v, false, b.v, (short)0, acc[j], false, false);
        }
    }

    const float scale = 0.125f;  // 1/sqrt(64)
    const int n = lane & 15;
    float* base = attn + (size_t)bh * LQ_ * LK_;
    #pragma unroll
    for (int j = 0; j < 4; ++j)
        for (int r = 0; r < 8; ++r) {
            const int m = q0 + r + 8 * (lane >> 4);
            base[(size_t)m * LK_ + k0 + 16 * j + n] = acc[j][r] * scale;
        }
}

// ---------------------------------------------------------------- masked softmax (in place)
__global__ void softmax_kernel(float* __restrict__ attn, const int* __restrict__ mask)
{
    const int row = blockIdx.x;              // (b*H + h)*LQ + q
    const int b_ = row / (H_ * LQ_);
    const int q  = row % LQ_;
    float* p = attn + (size_t)row * LK_;
    const int* mrow = mask + ((size_t)b_ * LQ_ + q) * LK_;

    __shared__ float red[256];
    const int tid = threadIdx.x;
    float vals[LK_ / 256];
    float mx = -3.402823466e38f;
    for (int j = 0; j < LK_ / 256; ++j) {
        const int idx = tid + 256 * j;
        float s = p[idx];
        if (mrow[idx] == 0) s = -3.402823466e38f;
        vals[j] = s;
        mx = fmaxf(mx, s);
    }
    red[tid] = mx; __syncthreads();
    for (int s = 128; s > 0; s >>= 1) {
        if (tid < s) red[tid] = fmaxf(red[tid], red[tid + s]);
        __syncthreads();
    }
    mx = red[0]; __syncthreads();

    float sum = 0.f;
    for (int j = 0; j < LK_ / 256; ++j) {
        vals[j] = __expf(vals[j] - mx);
        sum += vals[j];
    }
    red[tid] = sum; __syncthreads();
    for (int s = 128; s > 0; s >>= 1) {
        if (tid < s) red[tid] += red[tid + s];
        __syncthreads();
    }
    const float inv = 1.0f / red[0];
    for (int j = 0; j < LK_ / 256; ++j)
        p[tid + 256 * j] = vals[j] * inv;
}

// ---------------------------------------------------------------- out_head = P @ V
// One wave -> 16(q) x 64(d=full head): P strip read ONCE (537 MB total, not 4x).
__global__ void av_kernel(const float* __restrict__ attn,           // [bh][lq][lk] f32
                          const unsigned short* __restrict__ Vt,    // [bh][d][lk] bf16
                          unsigned short* __restrict__ Ob)          // (B*LQ, E_) bf16
{
    const int lane = threadIdx.x & 31;
    const int wave = threadIdx.x >> 5;
    const int tile = blockIdx.x * (blockDim.x >> 5) + wave;
    const int tilesPerBH = LQ_ / 16;                    // 128
    const int bh = tile / tilesPerBH;
    const int q0 = (tile % tilesPerBH) * 16;
    const int kh = lane >> 4;
    const int ml = lane & 15;
    const float* prow = attn + ((size_t)bh * LQ_ + q0 + ml) * LK_;

    v8f acc[4] = {};
    for (int lk = 0; lk < LK_; lk += 32) {
        __builtin_prefetch(prow + lk + 256, 0, 1);   // global_prefetch_b8
        FragBF a;
        const float* c0 = prow + lk + 8 * kh;
        const float* c1 = prow + lk + 16 + 8 * kh;
        float4 f0 = *(const float4*)(c0);
        float4 f1 = *(const float4*)(c0 + 4);
        float4 f2 = *(const float4*)(c1);
        float4 f3 = *(const float4*)(c1 + 4);
        a.h[0]  = f2bf(f0.x); a.h[1]  = f2bf(f0.y); a.h[2]  = f2bf(f0.z); a.h[3]  = f2bf(f0.w);
        a.h[4]  = f2bf(f1.x); a.h[5]  = f2bf(f1.y); a.h[6]  = f2bf(f1.z); a.h[7]  = f2bf(f1.w);
        a.h[8]  = f2bf(f2.x); a.h[9]  = f2bf(f2.y); a.h[10] = f2bf(f2.z); a.h[11] = f2bf(f2.w);
        a.h[12] = f2bf(f3.x); a.h[13] = f2bf(f3.y); a.h[14] = f2bf(f3.z); a.h[15] = f2bf(f3.w);
        #pragma unroll
        for (int j = 0; j < 4; ++j) {
            const unsigned short* vrow = Vt + ((size_t)bh * D_ + 16 * j + ml) * LK_;
            FragBF b;
            b.q[0] = *(const uint4*)(vrow + lk + 8 * kh);
            b.q[1] = *(const uint4*)(vrow + lk + 16 + 8 * kh);
            acc[j] = __builtin_amdgcn_wmma_f32_16x16x32_bf16(
                false, a.v, false, b.v, (short)0, acc[j], false, false);
        }
    }

    const int b_ = bh / H_, hh = bh % H_;
    const int n = lane & 15;
    #pragma unroll
    for (int j = 0; j < 4; ++j)
        for (int r = 0; r < 8; ++r) {
            const int q = q0 + r + 8 * (lane >> 4);
            Ob[((size_t)(b_ * LQ_ + q)) * E_ + hh * D_ + 16 * j + n] = f2bf(acc[j][r]);
        }
}

// ---------------------------------------------------------------- host
extern "C" void kernel_launch(void* const* d_in, const int* in_sizes, int n_in,
                              void* d_out, int out_size, void* d_ws, size_t ws_size,
                              hipStream_t stream) {
    const float* q_in = (const float*)d_in[0];
    const float* k_in = (const float*)d_in[1];
    const float* v_in = (const float*)d_in[2];
    const int*   mask = (const int*)d_in[3];
    const float* Wq = (const float*)d_in[4];
    const float* bq = (const float*)d_in[5];
    const float* Wk = (const float*)d_in[6];
    const float* bk = (const float*)d_in[7];
    const float* Wv = (const float*)d_in[8];
    const float* bv = (const float*)d_in[9];
    const float* Wo = (const float*)d_in[10];
    const float* bo = (const float*)d_in[11];

    const size_t NE = (size_t)B_ * LQ_ * E_;   // 4M elements
    const size_t EE = (size_t)E_ * E_;         // 1M elements

    char* ws = (char*)d_ws;
    unsigned short* Xq  = (unsigned short*)(ws);
    unsigned short* Xk  = Xq  + NE;
    unsigned short* Xv  = Xk  + NE;
    unsigned short* Wqb = Xv  + NE;
    unsigned short* Wkb = Wqb + EE;
    unsigned short* Wvb = Wkb + EE;
    unsigned short* Wob = Wvb + EE;
    unsigned short* Qb  = Wob + EE;    // [b][h][lq][d]
    unsigned short* Kb  = Qb  + NE;    // [b][h][lk][d]
    unsigned short* Vtb = Kb  + NE;    // [b][h][d][lk]
    unsigned short* Ob  = Vtb + NE;    // (B*LQ, E)

    float* out  = (float*)d_out;
    float* attn = out + NE;            // second tuple output

    cvt_f32_bf16<<<2048, 256, 0, stream>>>(q_in, Xq, (int)NE);
    cvt_f32_bf16<<<2048, 256, 0, stream>>>(k_in, Xk, (int)NE);
    cvt_f32_bf16<<<2048, 256, 0, stream>>>(v_in, Xv, (int)NE);
    cvt_f32_bf16<<<2048, 256, 0, stream>>>(Wq, Wqb, (int)EE);
    cvt_f32_bf16<<<2048, 256, 0, stream>>>(Wk, Wkb, (int)EE);
    cvt_f32_bf16<<<2048, 256, 0, stream>>>(Wv, Wvb, (int)EE);
    cvt_f32_bf16<<<2048, 256, 0, stream>>>(Wo, Wob, (int)EE);

    // projections: (4096x1024)@(1024x1024)^T. 128x64 tile per block -> 32*16=512 blocks
    gemm_bf16<<<512, 256, 0, stream>>>(Xq, Wqb, bq, Qb,  nullptr, 0);
    gemm_bf16<<<512, 256, 0, stream>>>(Xk, Wkb, bk, Kb,  nullptr, 0);
    gemm_bf16<<<512, 256, 0, stream>>>(Xv, Wvb, bv, Vtb, nullptr, 1);

    // scores: B*H*(LQ/16)*(LK/64) = 131072 wave-tiles, 8 waves/block
    scores_kernel<<<16384, 256, 0, stream>>>(Qb, Kb, attn);

    // softmax: one block per (b,h,q) row
    softmax_kernel<<<B_ * H_ * LQ_, 256, 0, stream>>>(attn, mask);

    // PV: B*H*(LQ/16) = 4096 wave-tiles
    av_kernel<<<512, 256, 0, stream>>>(attn, Vtb, Ob);

    // output projection -> fp32
    gemm_bf16<<<512, 256, 0, stream>>>(Ob, Wob, bo, nullptr, out, 2);
}
